// MambaClassifier_12618613916217
// MI455X (gfx1250) — compile-verified, matrix operand order checked by reading, not run on hardware
//
#include <hip/hip_runtime.h>

#define IN_DIM  64
#define D_MODEL 256
#define D_INNER 512
#define D_STATE 16
#define D_CONV  4
#define DT_RANK 16
#define N_LAYERS 4
#define BATCH   8
#define SEQ     1024
#define BL      (BATCH * SEQ)
#define XDBL_W  (DT_RANK + 2 * D_STATE)   // 48

typedef __attribute__((ext_vector_type(16))) __bf16 v16bf;
typedef __attribute__((ext_vector_type(8)))  float  v8f;

// ---------------------------------------------------------------------------
// Build one 16x32 bf16 WMMA fragment for a lane.
// CDNA5 16-bit A/B layout: lane (m = lane&15, h = lane>>4):
//   elements 0..7  hold K = k0 + 8h + i
//   elements 8..15 hold K = k0 + 16 + 8h + i
// Source is f32 row-major with K contiguous; convert to bf16 inline.
// ---------------------------------------------------------------------------
__device__ __forceinline__ v16bf mk_frag(const float* __restrict__ row, int k0, int h8) {
  v16bf f;
#pragma unroll
  for (int i = 0; i < 8; ++i) f[i] = (__bf16)row[k0 + h8 + i];
#pragma unroll
  for (int i = 0; i < 8; ++i) f[8 + i] = (__bf16)row[k0 + 16 + h8 + i];
  return f;
}

// ---------------------------------------------------------------------------
// C[m,n] = sum_k A[m,k] * W[n,k] (+ bias[n]).  A: MxK f32, W: NxK f32.
// One wave computes a 16x(16*NT) tile; A fragment reused NT times.
// Requires M%16==0, N%(16*NT)==0, K%32==0.
// ---------------------------------------------------------------------------
template <int NT>
__global__ void wmma_gemm_atn(const float* __restrict__ A,
                              const float* __restrict__ W,
                              const float* __restrict__ bias,
                              float* __restrict__ C,
                              int M, int N, int K)
{
  const int lane  = threadIdx.x & 31;
  const int wave  = (blockIdx.x * blockDim.x + threadIdx.x) >> 5;
  const int tilesM = M >> 4;
  const int groups = (N >> 4) / NT;
  const int tm = wave / groups;
  if (tm >= tilesM) return;               // wave-uniform: EXEC stays all-ones
  const int tn0 = (wave % groups) * NT;
  const int r15 = lane & 15;
  const int h8  = (lane >> 4) << 3;

  const float* arow = A + (size_t)(tm * 16 + r15) * K;
  const float* wrow[NT];
#pragma unroll
  for (int j = 0; j < NT; ++j)
    wrow[j] = W + (size_t)((tn0 + j) * 16 + r15) * K;

  v8f acc[NT] = {};
  for (int k0 = 0; k0 < K; k0 += 32) {
    v16bf a = mk_frag(arow, k0, h8);
#pragma unroll
    for (int j = 0; j < NT; ++j) {
      v16bf b = mk_frag(wrow[j], k0, h8);
      acc[j] = __builtin_amdgcn_wmma_f32_16x16x32_bf16(
                   false, a, false, b, (short)0, acc[j], false, false);
    }
  }

  // C/D layout: lane(0..15): N=lane, M=row r; lane(16..31): N=lane-16, M=8+r
  const int mb = tm * 16 + ((lane >> 4) << 3);
#pragma unroll
  for (int j = 0; j < NT; ++j) {
    const int n  = (tn0 + j) * 16 + r15;
    const float bv = bias ? bias[n] : 0.f;
#pragma unroll
    for (int r = 0; r < 8; ++r)
      C[(size_t)(mb + r) * N + n] = acc[j][r] + bv;
  }
}

// ---------------------------------------------------------------------------
// Depthwise causal conv (width 4) over L, + bias, + SiLU.
// xz is BLx(2*D_INNER); we read the first D_INNER columns (xh).
// ---------------------------------------------------------------------------
__global__ void conv_silu_kernel(const float* __restrict__ xz,
                                 const float* __restrict__ cw,
                                 const float* __restrict__ cb,
                                 float* __restrict__ xc)
{
  int idx = blockIdx.x * blockDim.x + threadIdx.x;
  if (idx >= BL * D_INNER) return;
  int d = idx & (D_INNER - 1);
  int t = (idx >> 9) & (SEQ - 1);
  int b = idx >> 19;
  float acc = cb[d];
#pragma unroll
  for (int k = 0; k < D_CONV; ++k) {
    int tt = t + k - (D_CONV - 1);
    if (tt >= 0)
      acc += xz[(size_t)(b * SEQ + tt) * (2 * D_INNER) + d] * cw[d * D_CONV + k];
  }
  xc[idx] = acc / (1.f + __expf(-acc));    // silu
}

// ---------------------------------------------------------------------------
// Fused selective scan. One lane per (batch, channel, state): 16 lanes
// cooperate per channel via __shfl_xor (DT_RANK == D_STATE == 16).
//   dt  = softplus(dtr . W_dt[d] + b_dt[d])          (16-lane dot)
//   h_n = exp(dt*A_n)*h_n + dt*B_n*xc
//   y   = (sum_n h_n*C_n + D*xc) * silu(z)
// ---------------------------------------------------------------------------
__global__ void scan_kernel(const float* __restrict__ xz,
                            const float* __restrict__ xc,
                            const float* __restrict__ xdb,
                            const float* __restrict__ Wdt,
                            const float* __restrict__ bdt,
                            const float* __restrict__ Alog,
                            const float* __restrict__ Dpp,
                            float* __restrict__ y)
{
  int gid = blockIdx.x * blockDim.x + threadIdx.x;
  int ch  = gid >> 4;
  if (ch >= BATCH * D_INNER) return;
  int n = gid & 15;
  int d = ch & (D_INNER - 1);
  int b = ch >> 9;

  float wdtn = Wdt[d * DT_RANK + n];
  float An   = -__expf(Alog[d * D_STATE + n]);
  float bd   = bdt[d];
  float Dd   = Dpp[d];
  float hn   = 0.f;

  for (int t = 0; t < SEQ; ++t) {
    int m = b * SEQ + t;
    const float* xr = xdb + (size_t)m * XDBL_W;
    // dt-projection: reduce 16 partial products across the lane group
    float dl = xr[n] * wdtn;
    dl += __shfl_xor(dl, 1);
    dl += __shfl_xor(dl, 2);
    dl += __shfl_xor(dl, 4);
    dl += __shfl_xor(dl, 8);
    dl += bd;
    float dt = (dl > 20.f) ? dl : log1pf(__expf(dl));  // softplus
    float xcv = xc[(size_t)m * D_INNER + d];
    hn = __expf(dt * An) * hn + dt * xr[DT_RANK + n] * xcv;
    float yp = hn * xr[DT_RANK + D_STATE + n];
    yp += __shfl_xor(yp, 1);
    yp += __shfl_xor(yp, 2);
    yp += __shfl_xor(yp, 4);
    yp += __shfl_xor(yp, 8);
    if (n == 0) {
      float zv = xz[(size_t)m * (2 * D_INNER) + D_INNER + d];
      float yv = yp + Dd * xcv;
      yv *= zv / (1.f + __expf(-zv));      // * silu(z)
      y[(size_t)m * D_INNER + d] = yv;
    }
  }
}

// ---------------------------------------------------------------------------
// Row LayerNorm over D_MODEL=256, one block (256 threads) per row.
// ---------------------------------------------------------------------------
__global__ void layernorm_kernel(const float* __restrict__ in,
                                 const float* __restrict__ g,
                                 const float* __restrict__ b,
                                 float* __restrict__ out)
{
  __shared__ float red[D_MODEL];
  int m = blockIdx.x, c = threadIdx.x;
  float v = in[(size_t)m * D_MODEL + c];
  red[c] = v;
  __syncthreads();
  for (int s = D_MODEL / 2; s > 0; s >>= 1) {
    if (c < s) red[c] += red[c + s];
    __syncthreads();
  }
  float mean = red[0] / D_MODEL;
  __syncthreads();
  float dv = v - mean;
  red[c] = dv * dv;
  __syncthreads();
  for (int s = D_MODEL / 2; s > 0; s >>= 1) {
    if (c < s) red[c] += red[c + s];
    __syncthreads();
  }
  float var = red[0] / D_MODEL;
  out[(size_t)m * D_MODEL + c] = dv * rsqrtf(var + 1e-5f) * g[c] + b[c];
}

// ---------------------------------------------------------------------------
// Mean-pool over SEQ then dot with Wf (1 x D_MODEL) + bf. One block per batch.
// ---------------------------------------------------------------------------
__global__ void pool_head_kernel(const float* __restrict__ h,
                                 const float* __restrict__ Wf,
                                 const float* __restrict__ bf,
                                 float* __restrict__ out)
{
  __shared__ float red[D_MODEL];
  int b = blockIdx.x, c = threadIdx.x;
  float acc = 0.f;
  for (int t = 0; t < SEQ; ++t)
    acc += h[(size_t)(b * SEQ + t) * D_MODEL + c];
  red[c] = (acc / SEQ) * Wf[c];
  __syncthreads();
  for (int s = D_MODEL / 2; s > 0; s >>= 1) {
    if (c < s) red[c] += red[c + s];
    __syncthreads();
  }
  if (c == 0) out[b] = red[0] + bf[0];
}

// ---------------------------------------------------------------------------
static inline void launch_gemm(const float* A, const float* W, const float* bias,
                               float* C, int M, int N, int K, hipStream_t s)
{
  int tm = M >> 4, tn = N >> 4;
  if ((tn & 3) == 0) {
    int waves = tm * (tn / 4);
    wmma_gemm_atn<4><<<(waves * 32 + 255) / 256, 256, 0, s>>>(A, W, bias, C, M, N, K);
  } else if (tn % 3 == 0) {
    int waves = tm * (tn / 3);
    wmma_gemm_atn<3><<<(waves * 32 + 255) / 256, 256, 0, s>>>(A, W, bias, C, M, N, K);
  } else {
    int waves = tm * tn;
    wmma_gemm_atn<1><<<(waves * 32 + 255) / 256, 256, 0, s>>>(A, W, bias, C, M, N, K);
  }
}

extern "C" void kernel_launch(void* const* d_in, const int* in_sizes, int n_in,
                              void* d_out, int out_size, void* d_ws, size_t ws_size,
                              hipStream_t stream)
{
  (void)in_sizes; (void)n_in; (void)out_size; (void)ws_size;
  const float* x      = (const float*)d_in[0];
  const float* Wp     = (const float*)d_in[1];
  const float* bp     = (const float*)d_in[2];
  const float* W_in   = (const float*)d_in[3];
  const float* conv_w = (const float*)d_in[4];
  const float* conv_b = (const float*)d_in[5];
  const float* W_x    = (const float*)d_in[6];
  const float* W_dt   = (const float*)d_in[7];
  const float* b_dt   = (const float*)d_in[8];
  const float* A_log  = (const float*)d_in[9];
  const float* Dpp    = (const float*)d_in[10];
  const float* W_out  = (const float*)d_in[11];
  const float* ln_g   = (const float*)d_in[12];
  const float* ln_b   = (const float*)d_in[13];
  const float* Wf     = (const float*)d_in[14];
  const float* bfp    = (const float*)d_in[15];
  float* out = (float*)d_out;

  float* ws   = (float*)d_ws;
  float* hbuf = ws;  ws += (size_t)BL * D_MODEL;        // 8 MB
  float* xz   = ws;  ws += (size_t)BL * 2 * D_INNER;    // 32 MB (also reused as W_out result)
  float* xc   = ws;  ws += (size_t)BL * D_INNER;        // 16 MB
  float* xdb  = ws;  ws += (size_t)BL * XDBL_W;         // 1.5 MB
  float* ybuf = ws;  ws += (size_t)BL * D_INNER;        // 16 MB
  float* obuf = xz;  // z no longer needed after scan -> reuse for W_out output

  // h = x @ Wp^T + bp
  launch_gemm(x, Wp, bp, hbuf, BL, D_MODEL, IN_DIM, stream);

  for (int l = 0; l < N_LAYERS; ++l) {
    launch_gemm(hbuf, W_in + (size_t)l * 2 * D_INNER * D_MODEL, nullptr,
                xz, BL, 2 * D_INNER, D_MODEL, stream);

    conv_silu_kernel<<<(BL * D_INNER + 255) / 256, 256, 0, stream>>>(
        xz, conv_w + (size_t)l * D_INNER * D_CONV, conv_b + (size_t)l * D_INNER, xc);

    launch_gemm(xc, W_x + (size_t)l * XDBL_W * D_INNER, nullptr,
                xdb, BL, XDBL_W, D_INNER, stream);

    scan_kernel<<<(BATCH * D_INNER * D_STATE + 255) / 256, 256, 0, stream>>>(
        xz, xc, xdb,
        W_dt + (size_t)l * D_INNER * DT_RANK,
        b_dt + (size_t)l * D_INNER,
        A_log + (size_t)l * D_INNER * D_STATE,
        Dpp + (size_t)l * D_INNER, ybuf);

    launch_gemm(ybuf, W_out + (size_t)l * D_MODEL * D_INNER, nullptr,
                obuf, BL, D_MODEL, D_INNER, stream);

    layernorm_kernel<<<BL, D_MODEL, 0, stream>>>(
        obuf, ln_g + (size_t)l * D_MODEL, ln_b + (size_t)l * D_MODEL, hbuf);
  }

  pool_head_kernel<<<BATCH, D_MODEL, 0, stream>>>(hbuf, Wf, bfp, out);
}
